// Pointnet_Plus_Plus_33449205301355
// MI455X (gfx1250) — matrix operand order repaired and checked by reading, not compiled, not run on hardware
//
#include <hip/hip_runtime.h>
#include <hip/hip_bf16.h>

typedef __attribute__((ext_vector_type(16))) __bf16 v16bf;
typedef __attribute__((ext_vector_type(8)))  float  v8f;

// ---------------- workspace layout (bytes) ----------------
// A: bf16 activations (GEMM inputs / post-BN acts)   262144*128*2 = 67,108,864
// B: bf16 GEMM outputs (pre-BN)                      262144*128*2 = 67,108,864
// f1: fp32 (4,2048,128)  f2: fp32 (4,512,256)  f3: fp32 (4,128,256)
// stats: 512 floats (sum, sumsq per channel)
// wts: bf16 padded weights, 288768 elements
#define WS_A_OFF   0ull
#define WS_B_OFF   67108864ull
#define WS_F1_OFF  134217728ull
#define WS_F2_OFF  138412032ull
#define WS_F3_OFF  140509184ull
#define WS_ST_OFF  141033472ull
#define WS_WT_OFF  141035520ull

// ---------------- weight prep: fp32 (cout,cin) -> bf16 (cout,kpad) zero-padded ----
__global__ void k_prep_w(const float* __restrict__ w, __bf16* __restrict__ wout,
                         int cin, int cout, int kpad) {
  int i = blockIdx.x * blockDim.x + threadIdx.x;
  if (i >= cout * kpad) return;
  int o = i / kpad, k = i % kpad;
  float v = (k < cin) ? w[o * cin + k] : 0.0f;
  wout[i] = (__bf16)v;
}

__global__ void k_zero_stats(float* __restrict__ s) {
  int i = threadIdx.x;
  if (i < 512) s[i] = 0.0f;
}

// ---------------- ball grouping: one wave32 per query point ----------------
// Queries are prefixes of the pointcloud; gather first 32 in-radius indices
// (ascending), replicate first (or 0) to fill, write bf16 row [coords|feats|0pad].
__global__ void k_ball_group(const float* __restrict__ pc,
                             const float* __restrict__ feats,
                             __bf16* __restrict__ X,
                             int S, int N, int Cf, int Kpad, float r2) {
  int b = blockIdx.x / S;
  int s = blockIdx.x % S;
  int lane = threadIdx.x;                   // 0..31
  const float* base = pc + (size_t)b * 8192 * 3;
  float cx = base[s * 3 + 0], cy = base[s * 3 + 1], cz = base[s * 3 + 2];

  __shared__ int idxbuf[32];
  int count = 0;
  for (int nb = 0; nb < N && count < 32; nb += 32) {
    int n = nb + lane;
    float dx = base[n * 3 + 0] - cx;
    float dy = base[n * 3 + 1] - cy;
    float dz = base[n * 3 + 2] - cz;
    bool in = (dx * dx + dy * dy + dz * dz) < r2;
    unsigned msk = __builtin_amdgcn_ballot_w32(in);
    int pos = count + __builtin_popcount(msk & ((1u << lane) - 1u));
    if (in && pos < 32) idxbuf[pos] = n;
    count += __builtin_popcount(msk);
  }
  __syncthreads();
  int total = count < 32 ? count : 32;
  int first = (total > 0) ? idxbuf[0] : 0;
  int myn   = (lane < total) ? idxbuf[lane] : first;

  __bf16* row = X + ((size_t)blockIdx.x * 32 + lane) * Kpad;
  row[0] = (__bf16)(base[myn * 3 + 0] - cx);
  row[1] = (__bf16)(base[myn * 3 + 1] - cy);
  row[2] = (__bf16)(base[myn * 3 + 2] - cz);
  if (Cf > 0) {
    const float* f = feats + ((size_t)b * N + myn) * Cf;
    for (int c = 0; c < Cf; ++c) row[3 + c] = (__bf16)f[c];
  }
  for (int k = 3 + Cf; k < Kpad; ++k) row[k] = (__bf16)0.0f;
}

// ---------------- bf16 WMMA GEMM: Y = X(MxK) * W^T(KxC) + bias ----------------
// One wave per 16x64 tile (4 accumulators): A fragment loaded once per K-chunk
// and reused by 4 WMMAs. Epilogue: bf16 store + fp32 atomic sum/sumsq stats.
__global__ void k_gemm(const __bf16* __restrict__ X, const __bf16* __restrict__ W,
                       const float* __restrict__ bias, __bf16* __restrict__ Y,
                       float* __restrict__ stats, int M, int Kpad, int Cout) {
  int lane = threadIdx.x;
  int half = lane >> 4;
  int m    = lane & 15;
  int row0 = blockIdx.x * 16;
  int col0 = blockIdx.y * 64;

  const __bf16* xrow = X + (size_t)(row0 + m) * Kpad;
  const __bf16* wrow = W + (size_t)(col0 + m) * Kpad;
  size_t wstep = (size_t)16 * Kpad;          // stride between column sub-tiles

  v8f acc0 = {}, acc1 = {}, acc2 = {}, acc3 = {};
  for (int kb = 0; kb < Kpad; kb += 32) {
    v16bf a;
    const __bf16* xa = xrow + kb + 16 * half; // A 16x32: 16 contiguous K
#pragma unroll
    for (int i = 0; i < 8; ++i) { a[i] = xa[i]; a[i + 8] = xa[i + 8]; }

    const __bf16* wb0 = wrow + kb + 16 * half; // B 32x16: 16 contiguous K
    const __bf16* wb1 = wb0 + wstep;
    const __bf16* wb2 = wb1 + wstep;
    const __bf16* wb3 = wb2 + wstep;
    v16bf b0, b1, b2, b3;
#pragma unroll
    for (int i = 0; i < 16; ++i) b0[i] = wb0[i];
#pragma unroll
    for (int i = 0; i < 16; ++i) b1[i] = wb1[i];
#pragma unroll
    for (int i = 0; i < 16; ++i) b2[i] = wb2[i];
#pragma unroll
    for (int i = 0; i < 16; ++i) b3[i] = wb3[i];

    acc0 = __builtin_amdgcn_wmma_f32_16x16x32_bf16(false, a, false, b0, (short)0, acc0, false, false);
    acc1 = __builtin_amdgcn_wmma_f32_16x16x32_bf16(false, a, false, b1, (short)0, acc1, false, false);
    acc2 = __builtin_amdgcn_wmma_f32_16x16x32_bf16(false, a, false, b2, (short)0, acc2, false, false);
    acc3 = __builtin_amdgcn_wmma_f32_16x16x32_bf16(false, a, false, b3, (short)0, acc3, false, false);
  }

#define EPILOGUE(T, ACC)                                                       \
  {                                                                            \
    int col = col0 + (T) * 16 + m;                                             \
    float bia = bias[col];                                                     \
    float s1 = 0.0f, s2 = 0.0f;                                                \
    _Pragma("unroll") for (int r = 0; r < 8; ++r) {                            \
      float v = (ACC)[r] + bia;                                                \
      int row = row0 + r + 8 * half;                                           \
      Y[(size_t)row * Cout + col] = (__bf16)v;                                 \
      s1 += v;                                                                 \
      s2 += v * v;                                                             \
    }                                                                          \
    atomicAdd(&stats[col], s1);                                                \
    atomicAdd(&stats[256 + col], s2);                                          \
  }
  EPILOGUE(0, acc0)
  EPILOGUE(1, acc1)
  EPILOGUE(2, acc2)
  EPILOGUE(3, acc3)
#undef EPILOGUE
}

// ---------------- BN (global mean/var over M) + ReLU, write bf16 act -----------
__global__ void k_bn_relu(const __bf16* __restrict__ Y, const float* __restrict__ stats,
                          const float* __restrict__ g, const float* __restrict__ h,
                          __bf16* __restrict__ A, int M, int C) {
  size_t i = (size_t)blockIdx.x * blockDim.x + threadIdx.x;
  size_t total = (size_t)M * C;
  if (i >= total) return;
  int c = (int)(i % C);
  float invM = 1.0f / (float)M;
  float mean = stats[c] * invM;
  float var  = stats[256 + c] * invM - mean * mean;
  float inv  = rsqrtf(var + 1e-5f);
  float y = (float)Y[i];
  float v = (y - mean) * inv * g[c] + h[c];
  A[i] = (__bf16)(v > 0.0f ? v : 0.0f);
}

// ---------------- max over 32 neighbors -> fp32 feats -------------------------
__global__ void k_maxpool(const __bf16* __restrict__ A, float* __restrict__ f,
                          int P, int C) {
  int i = blockIdx.x * blockDim.x + threadIdx.x;
  if (i >= P * C) return;
  int p = i / C, c = i % C;
  float mx = -3.4e38f;
#pragma unroll 4
  for (int j = 0; j < 32; ++j) {
    float v = (float)A[((size_t)p * 32 + j) * C + c];
    mx = fmaxf(mx, v);
  }
  f[i] = mx;
}

// ---------------- pack outputs: (xyz3, feats3, idx0) ---------------------------
__global__ void k_outputs(const float* __restrict__ pc, const float* __restrict__ f3,
                          float* __restrict__ out) {
  int i = blockIdx.x * blockDim.x + threadIdx.x;
  if (i >= 140800) return;
  if (i < 1536) {                       // xyz3: (4,128,3) = first 128 pts/batch
    int b = i / 384, r = i % 384;
    int s = r / 3, d = r % 3;
    out[i] = pc[((size_t)b * 8192 + s) * 3 + d];
  } else if (i < 1536 + 131072) {       // feats3: (4,128,256)
    int j = i - 1536;
    out[i] = f3[j];
  } else {                              // idx0: (4,2048) = tiled arange
    int j = i - 132608;
    int s = j % 2048;
    out[i] = (float)s;
  }
}

extern "C" void kernel_launch(void* const* d_in, const int* in_sizes, int n_in,
                              void* d_out, int out_size, void* d_ws, size_t ws_size,
                              hipStream_t stream) {
  const float* pc = (const float*)d_in[0];
  const float *W[9], *Bs[9], *G[9], *H[9];
  for (int i = 0; i < 9; ++i) {
    W[i]  = (const float*)d_in[1 + 4 * i];
    Bs[i] = (const float*)d_in[2 + 4 * i];
    G[i]  = (const float*)d_in[3 + 4 * i];
    H[i]  = (const float*)d_in[4 + 4 * i];
  }
  char* ws = (char*)d_ws;
  __bf16* A    = (__bf16*)(ws + WS_A_OFF);
  __bf16* Bbuf = (__bf16*)(ws + WS_B_OFF);
  float*  f1   = (float*)(ws + WS_F1_OFF);
  float*  f2   = (float*)(ws + WS_F2_OFF);
  float*  f3   = (float*)(ws + WS_F3_OFF);
  float*  st   = (float*)(ws + WS_ST_OFF);
  __bf16* wb   = (__bf16*)(ws + WS_WT_OFF);

  static const int cin[9]  = {3, 64, 64, 131, 128, 128, 259, 256, 256};
  static const int cout[9] = {64, 64, 128, 128, 128, 256, 256, 256, 256};
  static const int kpad[9] = {32, 64, 64, 160, 128, 128, 288, 256, 256};
  size_t woff[9]; size_t acc = 0;
  for (int i = 0; i < 9; ++i) { woff[i] = acc; acc += (size_t)cout[i] * kpad[i]; }

  for (int i = 0; i < 9; ++i) {
    int n = cout[i] * kpad[i];
    k_prep_w<<<(n + 255) / 256, 256, 0, stream>>>(W[i], wb + woff[i], cin[i], cout[i], kpad[i]);
  }

  const int   Sl[3] = {2048, 512, 128};
  const int   Nl[3] = {8192, 2048, 512};
  const int   Cfl[3] = {0, 128, 256};
  const float r2l[3] = {0.09f, 0.25f, 0.49f};
  const float* featin[3] = {nullptr, f1, f2};
  float* featout[3] = {f1, f2, f3};

  for (int li = 0; li < 3; ++li) {
    int S = Sl[li];
    int M = 4 * S * 32;
    int m0 = 3 * li;
    k_ball_group<<<4 * S, 32, 0, stream>>>(pc, featin[li], A,
                                           S, Nl[li], Cfl[li], kpad[m0], r2l[li]);
    for (int j = 0; j < 3; ++j) {
      int mi = m0 + j;
      int K = kpad[mi], C = cout[mi];
      k_zero_stats<<<1, 512, 0, stream>>>(st);
      dim3 grid(M / 16, C / 64);
      k_gemm<<<grid, 32, 0, stream>>>(A, wb + woff[mi], Bs[mi], Bbuf, st, M, K, C);
      size_t tot = (size_t)M * C;
      k_bn_relu<<<(unsigned)((tot + 255) / 256), 256, 0, stream>>>(Bbuf, st, G[mi], H[mi], A, M, C);
    }
    int Cl = cout[m0 + 2];
    int n = 4 * S * Cl;
    k_maxpool<<<(n + 255) / 256, 256, 0, stream>>>(A, featout[li], 4 * S, Cl);
  }

  k_outputs<<<(140800 + 255) / 256, 256, 0, stream>>>(pc, f3, (float*)d_out);
}
